// WindowAttention_46523085750724
// MI455X (gfx1250) — matrix operand order repaired and checked by reading, not compiled, not run on hardware
//
#include <hip/hip_runtime.h>

typedef __attribute__((ext_vector_type(16))) _Float16 v16h;
typedef __attribute__((ext_vector_type(8)))  float    v8f;
typedef __attribute__((ext_vector_type(2)))  _Float16 h2;

#define NWIN 49
#define NPAD 64
#define HD   32
#define NH   3
#define CCH  96
#define NEG_BIG (-1e30f)

// XOR-lane swap via DS_SWIZZLE_B32 (group-of-32 mode):
// offset[4:0]=and_mask=0x1f, offset[9:5]=or_mask=0, offset[14:10]=xor_mask.
// EXEC is all-ones in the compute phase, so this matches shfl_xor exactly
// without the bpermute address-math / bounds-check VALU overhead.
template <int XORMASK>
__device__ __forceinline__ float swz_xor(float x) {
  int r = __builtin_amdgcn_ds_swizzle(__float_as_int(x),
                                      0x1f | (XORMASK << 10));
  return __int_as_float(r);
}

// ---------------------------------------------------------------------------
// Kernel 1: precompute the MLP relative-position bias table, padded to 64x64
// with -1e30 in masked columns/rows (drives softmax masking for free).
// bt layout: [3][64][64] f32 in workspace.
// ---------------------------------------------------------------------------
__global__ __launch_bounds__(256) void bias_table_kernel(
    const float* __restrict__ rel, const float* __restrict__ w1,
    const float* __restrict__ b1,  const float* __restrict__ w2,
    const float* __restrict__ b2,  float* __restrict__ bt) {
  int idx = blockIdx.x * blockDim.x + threadIdx.x;
  if (idx >= NPAD * NPAD) return;
  int n = idx >> 6, m = idx & 63;
  float a0, a1, a2;
  if (n < NWIN && m < NWIN) {
    float x0 = rel[(n * NWIN + m) * 2 + 0];
    float x1 = rel[(n * NWIN + m) * 2 + 1];
    a0 = b2[0]; a1 = b2[1]; a2 = b2[2];
    for (int j = 0; j < 256; ++j) {
      float hv = fmaxf(0.0f, x0 * w1[j] + x1 * w1[256 + j] + b1[j]);
      a0 += hv * w2[j * 3 + 0];
      a1 += hv * w2[j * 3 + 1];
      a2 += hv * w2[j * 3 + 2];
    }
  } else {
    a0 = a1 = a2 = NEG_BIG;   // masks padded rows/cols through softmax
  }
  bt[0 * NPAD * NPAD + idx] = a0;
  bt[1 * NPAD * NPAD + idx] = a1;
  bt[2 * NPAD * NPAD + idx] = a2;
}

// ---------------------------------------------------------------------------
// Kernel 2: fused window attention. One block per window, 12 waves:
// wave w -> (head h = w/4, row-tile nt = w%4). Both GEMMs via
// v_wmma_f32_16x16x32_f16; softmax in fp32 registers with ds_swizzle
// 16-lane reductions.
// ---------------------------------------------------------------------------
__global__ __launch_bounds__(384) void win_attn_kernel(
    const float* __restrict__ q, const float* __restrict__ k,
    const float* __restrict__ v, const float* __restrict__ bias,
    float* __restrict__ out) {
  __shared__ _Float16 qs[NH][NPAD][HD];     // 12 KB, scale folded in
  __shared__ _Float16 ks[NH][NPAD][HD];     // 12 KB
  __shared__ _Float16 vts[NH][HD][NPAD];    // 12 KB, V transposed (d-major)
  __shared__ _Float16 ps[12][16][NPAD];     // 24 KB, per-wave P staging

  const int tid = threadIdx.x;
  const size_t base = (size_t)blockIdx.x * (NWIN * CCH);
  const float scale = 0.17677669529663687f; // 32^-0.5

  // ---- cooperative load: one HBM pass, float4 vectorized ----
  for (int i4 = tid; i4 < (NWIN * CCH) / 4; i4 += blockDim.x) {
    int n  = i4 / (CCH / 4);
    int c4 = (i4 % (CCH / 4)) * 4;
    int hh = c4 / HD;
    int d  = c4 % HD;
    const float4 qv = *(const float4*)(q + base + n * CCH + c4);
    const float4 kv = *(const float4*)(k + base + n * CCH + c4);
    const float4 vv = *(const float4*)(v + base + n * CCH + c4);
    qs[hh][n][d + 0] = (_Float16)(qv.x * scale);
    qs[hh][n][d + 1] = (_Float16)(qv.y * scale);
    qs[hh][n][d + 2] = (_Float16)(qv.z * scale);
    qs[hh][n][d + 3] = (_Float16)(qv.w * scale);
    ks[hh][n][d + 0] = (_Float16)kv.x;
    ks[hh][n][d + 1] = (_Float16)kv.y;
    ks[hh][n][d + 2] = (_Float16)kv.z;
    ks[hh][n][d + 3] = (_Float16)kv.w;
    vts[hh][d + 0][n] = (_Float16)vv.x;
    vts[hh][d + 1][n] = (_Float16)vv.y;
    vts[hh][d + 2][n] = (_Float16)vv.z;
    vts[hh][d + 3][n] = (_Float16)vv.w;
  }
  // zero the padded rows n in [49,64)
  for (int i = tid; i < NH * (NPAD - NWIN) * HD; i += blockDim.x) {
    int hh = i / ((NPAD - NWIN) * HD);
    int rr = i % ((NPAD - NWIN) * HD);
    int n  = NWIN + rr / HD;
    int d  = rr % HD;
    qs[hh][n][d]  = (_Float16)0.0f;
    ks[hh][n][d]  = (_Float16)0.0f;
    vts[hh][d][n] = (_Float16)0.0f;
  }
  __syncthreads();

  const int lane = tid & 31;
  const int w    = tid >> 5;
  const int h    = w >> 2;      // head
  const int nt   = w & 3;       // 16-row tile of padded N
  const int ln   = lane & 15;
  const int kh   = lane >> 4;   // lane half -> K-group / M-group select

  // ---- A = Q tile [16x32] f16, documented A-layout ----
  v16h aQ;
#pragma unroll
  for (int vv2 = 0; vv2 < 8; ++vv2) {
    int kb = ((vv2 & 3) * 2) + ((vv2 >> 2) * 16) + kh * 8;
    h2 pr = *(const h2*)&qs[h][nt * 16 + ln][kb];
    aQ[2 * vv2] = pr.x; aQ[2 * vv2 + 1] = pr.y;
  }

  // ---- S = Q Kt : 4 WMMAs (full K=hd=32 each) ----
  v8f ct[4];
#pragma unroll
  for (int t = 0; t < 4; ++t) {
    v16h bK;  // B layout: lane = column m, K striped; contiguous in K rows
#pragma unroll
    for (int vv2 = 0; vv2 < 8; ++vv2) {
      h2 pr = *(const h2*)&ks[h][t * 16 + ln][((vv2 & 3) * 2) + ((vv2 >> 2) * 16) + kh * 8];
      bK[2 * vv2] = pr.x; bK[2 * vv2 + 1] = pr.y;
    }
    v8f cz = {};
    ct[t] = __builtin_amdgcn_wmma_f32_16x16x32_f16(
        false, aQ, false, bK, (short)0, cz, false, false);
  }

  // ---- add bias (padded columns carry -1e30 -> masked) ----
#pragma unroll
  for (int t = 0; t < 4; ++t) {
#pragma unroll
    for (int r = 0; r < 8; ++r) {
      int n = nt * 16 + r + 8 * kh;
      int m = t * 16 + ln;
      ct[t][r] += bias[(h * NPAD + n) * NPAD + m];
    }
  }

  // ---- row softmax: 16-lane ds_swizzle reductions (rows live in halves) ----
#pragma unroll
  for (int r = 0; r < 8; ++r) {
    float rmax = fmaxf(fmaxf(ct[0][r], ct[1][r]), fmaxf(ct[2][r], ct[3][r]));
    rmax = fmaxf(rmax, swz_xor<1>(rmax));
    rmax = fmaxf(rmax, swz_xor<2>(rmax));
    rmax = fmaxf(rmax, swz_xor<4>(rmax));
    rmax = fmaxf(rmax, swz_xor<8>(rmax));
    float rsum = 0.0f;
#pragma unroll
    for (int t = 0; t < 4; ++t) {
      float e = __expf(ct[t][r] - rmax);
      ct[t][r] = e;
      rsum += e;
    }
    rsum += swz_xor<1>(rsum);
    rsum += swz_xor<2>(rsum);
    rsum += swz_xor<4>(rsum);
    rsum += swz_xor<8>(rsum);
    float inv = 1.0f / rsum;
#pragma unroll
    for (int t = 0; t < 4; ++t)
      ps[w][r + 8 * kh][t * 16 + ln] = (_Float16)(ct[t][r] * inv);
  }

  // wave-private C-layout -> A-layout transpose bounce through LDS
  asm volatile("s_wait_dscnt 0" ::: "memory");

  v16h aP0, aP1;  // P[:,0:32] and P[:,32:64]
#pragma unroll
  for (int vv2 = 0; vv2 < 8; ++vv2) {
    int kb = ((vv2 & 3) * 2) + ((vv2 >> 2) * 16) + kh * 8;
    h2 p0 = *(const h2*)&ps[w][ln][kb];
    h2 p1 = *(const h2*)&ps[w][ln][kb + 32];
    aP0[2 * vv2] = p0.x; aP0[2 * vv2 + 1] = p0.y;
    aP1[2 * vv2] = p1.x; aP1[2 * vv2 + 1] = p1.y;
  }

  // ---- O = P V : V transposed in LDS so B loads are contiguous pairs ----
#pragma unroll
  for (int dt = 0; dt < 2; ++dt) {
    v16h bV0, bV1;
#pragma unroll
    for (int vv2 = 0; vv2 < 8; ++vv2) {
      h2 q0 = *(const h2*)&vts[h][dt * 16 + ln][((vv2 & 3) * 2) + ((vv2 >> 2) * 16) + kh * 8];
      h2 q1 = *(const h2*)&vts[h][dt * 16 + ln][32 + ((vv2 & 3) * 2) + ((vv2 >> 2) * 16) + kh * 8];
      bV0[2 * vv2] = q0.x; bV0[2 * vv2 + 1] = q0.y;
      bV1[2 * vv2] = q1.x; bV1[2 * vv2 + 1] = q1.y;
    }
    v8f o = {};
    o = __builtin_amdgcn_wmma_f32_16x16x32_f16(
        false, aP0, false, bV0, (short)0, o, false, false);
    o = __builtin_amdgcn_wmma_f32_16x16x32_f16(
        false, aP1, false, bV1, (short)0, o, false, false);
#pragma unroll
    for (int r = 0; r < 8; ++r) {
      int n = nt * 16 + r + 8 * kh;
      if (n < NWIN)
        out[base + n * CCH + h * HD + dt * 16 + ln] = o[r];
    }
  }
}

// ---------------------------------------------------------------------------
extern "C" void kernel_launch(void* const* d_in, const int* in_sizes, int n_in,
                              void* d_out, int out_size, void* d_ws, size_t ws_size,
                              hipStream_t stream) {
  const float* q   = (const float*)d_in[0];
  const float* k   = (const float*)d_in[1];
  const float* v   = (const float*)d_in[2];
  const float* rel = (const float*)d_in[3];
  const float* w1  = (const float*)d_in[4];
  const float* b1  = (const float*)d_in[5];
  const float* w2  = (const float*)d_in[6];
  const float* b2  = (const float*)d_in[7];
  float* out = (float*)d_out;
  float* biasTab = (float*)d_ws;   // [3][64][64] f32 = 48 KB

  const int B = in_sizes[0] / (NWIN * CCH);

  bias_table_kernel<<<(NPAD * NPAD + 255) / 256, 256, 0, stream>>>(
      rel, w1, b1, w2, b2, biasTab);
  win_attn_kernel<<<B, 384, 0, stream>>>(q, k, v, biasTab, out);
}
